// ResidualConvBlock_84447646974225
// MI455X (gfx1250) — matrix operand order repaired and checked by reading, not compiled, not run on hardware
//
#include <hip/hip_runtime.h>

typedef __attribute__((ext_vector_type(16))) __bf16 v16bf;
typedef __attribute__((ext_vector_type(4)))  __bf16 v4bf;
typedef __attribute__((ext_vector_type(8)))  float  v8f;

#define IN_F  128
#define OUT_F 256

// ---------------------------------------------------------------------------
// WMMA helpers (CDNA5 16x16x32 bf16, f32 accumulate)
// A frag layout (ISA 7.12.2, 16-bit A 16x32): lane L<16 -> row L, K halves
// [kbase..kbase+7] in elems 0..7 and [kbase+16..kbase+23] in elems 8..15;
// lanes 16..31 same rows, K offset +8.
// B frag layout: lane L<16 -> col L, K = kbase+0..7 in elems 0..7, kbase+16..23 in elems 8..15; lanes 16..31
// -> same cols, K offset +8 (same order as A).  We pre-pack weights so each lane's 16
// bf16 B values are contiguous (one 32B load).
// ---------------------------------------------------------------------------
__device__ __forceinline__ v16bf load_a_frag(const __bf16* p) {
  v16bf a;
#pragma unroll
  for (int i = 0; i < 8; ++i) { a[i] = p[i]; a[i + 8] = p[i + 16]; }
  return a;
}

__device__ __forceinline__ v8f wmma_bf16(v16bf a, v16bf b, v8f c) {
  return __builtin_amdgcn_wmma_f32_16x16x32_bf16(false, a, false, b,
                                                 (short)0, c, false, false);
}

// ---------------------------------------------------------------------------
// Pack fp32 weight [K][256] into bf16 fragment-major:
//   Wp[((k/32)*256 + n)*32 + (k%32 with bits 3,4 swapped)] = bf16(W[k][n])
// ---------------------------------------------------------------------------
__global__ void k_pack_w(const float* __restrict__ W, __bf16* __restrict__ Wp, int K) {
  int idx = blockIdx.x * blockDim.x + threadIdx.x;
  if (idx >= K * OUT_F) return;
  int k = idx >> 8;          // row (K dim)
  int n = idx & 255;         // col
  Wp[(((size_t)(k >> 5) * OUT_F + n) << 5) + ((k & 7) | ((k & 8) << 1) | ((k & 16) >> 1))] = (__bf16)W[idx];
}

// ---------------------------------------------------------------------------
// LayerNorm over 128 feats, one wave per row; writes f32 + bf16 copies.
// ---------------------------------------------------------------------------
__global__ void k_ln1(const float* __restrict__ h, const float* __restrict__ g,
                      const float* __restrict__ bta, float* __restrict__ of,
                      __bf16* __restrict__ ob, int N) {
  int row  = (blockIdx.x * blockDim.x + threadIdx.x) >> 5;
  int lane = threadIdx.x & 31;
  if (row >= N) return;
  const float4 v = *(const float4*)(h + (size_t)row * IN_F + lane * 4);
  float s  = v.x + v.y + v.z + v.w;
  float ss = v.x * v.x + v.y * v.y + v.z * v.z + v.w * v.w;
#pragma unroll
  for (int m = 16; m > 0; m >>= 1) {
    s  += __shfl_xor(s, m, 32);
    ss += __shfl_xor(ss, m, 32);
  }
  float mu = s * (1.0f / IN_F);
  float rs = rsqrtf(ss * (1.0f / IN_F) - mu * mu + 1e-5f);
  const float4 gv = *(const float4*)(g + lane * 4);
  const float4 bv = *(const float4*)(bta + lane * 4);
  float4 o;
  o.x = (v.x - mu) * rs * gv.x + bv.x;
  o.y = (v.y - mu) * rs * gv.y + bv.y;
  o.z = (v.z - mu) * rs * gv.z + bv.z;
  o.w = (v.w - mu) * rs * gv.w + bv.w;
  *(float4*)(of + (size_t)row * IN_F + lane * 4) = o;
  v4bf p = { (__bf16)o.x, (__bf16)o.y, (__bf16)o.z, (__bf16)o.w };
  *(v4bf*)(ob + (size_t)row * IN_F + lane * 4) = p;
}

// ---------------------------------------------------------------------------
// Edge scatter: one wave per edge; gather 128 f32 feats of src, atomic-add
// into agg[dst]; lane 0 bumps degree.
// ---------------------------------------------------------------------------
__global__ void k_scatter(const float* __restrict__ hn, const int* __restrict__ src,
                          const int* __restrict__ dst, float* __restrict__ agg,
                          float* __restrict__ deg, int E) {
  int e    = (blockIdx.x * blockDim.x + threadIdx.x) >> 5;
  int lane = threadIdx.x & 31;
  if (e >= E) return;
  int s = src[e];
  int d = dst[e];
  const float4 v = *(const float4*)(hn + (size_t)s * IN_F + lane * 4);
  float* ap = agg + (size_t)d * IN_F + lane * 4;
  atomicAdd(ap + 0, v.x);
  atomicAdd(ap + 1, v.y);
  atomicAdd(ap + 2, v.z);
  atomicAdd(ap + 3, v.w);
  if (lane == 0) atomicAdd(deg + d, 1.0f);
}

// ---------------------------------------------------------------------------
// mean_neigh = agg / max(deg,1), cast to bf16.  32 threads per node.
// ---------------------------------------------------------------------------
__global__ void k_mean(const float* __restrict__ agg, const float* __restrict__ deg,
                       __bf16* __restrict__ mean, int N) {
  int idx = blockIdx.x * blockDim.x + threadIdx.x;
  if (idx >= N * 32) return;
  int node = idx >> 5;
  int q    = idx & 31;
  float dv  = deg[node];
  float inv = 1.0f / (dv > 1.0f ? dv : 1.0f);
  const float4 a = *(const float4*)(agg + (size_t)node * IN_F + q * 4);
  v4bf m = { (__bf16)(a.x * inv), (__bf16)(a.y * inv),
             (__bf16)(a.z * inv), (__bf16)(a.w * inv) };
  *(v4bf*)(mean + (size_t)node * IN_F + q * 4) = m;
}

// ---------------------------------------------------------------------------
// GEMM1: conv = hn_bf16 @ W_self + mean_bf16 @ W_neigh + b + repeat(h_skip,2)
// Block = 16 rows x 256 cols, 8 waves; wave w covers cols [32w,32w+32).
// ---------------------------------------------------------------------------
__global__ void k_gemm1(const __bf16* __restrict__ A1, const __bf16* __restrict__ A2,
                        const __bf16* __restrict__ W1, const __bf16* __restrict__ W2,
                        const float* __restrict__ bias, const float* __restrict__ hskip,
                        float* __restrict__ conv, int N) {
  const int LDA = IN_F + 8;  // 136, pad to dodge bank conflicts
  __shared__ __bf16 sA1[16 * (IN_F + 8)];
  __shared__ __bf16 sA2[16 * (IN_F + 8)];
  const int tid = threadIdx.x;
  const int rowbase = blockIdx.x * 16;
  {
    int r  = tid >> 4;
    int c  = (tid & 15) * 8;
    int gr = rowbase + r; if (gr >= N) gr = N - 1;
    *(uint4*)&sA1[r * LDA + c] = *(const uint4*)(A1 + (size_t)gr * IN_F + c);
    *(uint4*)&sA2[r * LDA + c] = *(const uint4*)(A2 + (size_t)gr * IN_F + c);
  }
  __syncthreads();
  const int lane    = tid & 31;
  const int colbase = (tid >> 5) * 32;
  const int arow  = lane & 15;
  const int khalf = (lane >> 4) * 8;
  const int bn    = lane & 15;
  const int bk    = (lane >> 4) * 16;
  v8f acc0 = {};
  v8f acc1 = {};
#pragma unroll
  for (int kt = 0; kt < IN_F / 32; ++kt) {
    v16bf a1 = load_a_frag(&sA1[arow * LDA + kt * 32 + khalf]);
    v16bf a2 = load_a_frag(&sA2[arow * LDA + kt * 32 + khalf]);
    const __bf16* w1 = W1 + (((size_t)kt * OUT_F + colbase + bn) << 5) + bk;
    const __bf16* w2 = W2 + (((size_t)kt * OUT_F + colbase + bn) << 5) + bk;
    acc0 = wmma_bf16(a1, *(const v16bf*)w1, acc0);
    acc0 = wmma_bf16(a2, *(const v16bf*)w2, acc0);
    acc1 = wmma_bf16(a1, *(const v16bf*)(w1 + (16 << 5)), acc1);
    acc1 = wmma_bf16(a2, *(const v16bf*)(w2 + (16 << 5)), acc1);
  }
  const int mb = (lane >> 4) * 8;
  const int n  = lane & 15;
#pragma unroll
  for (int i = 0; i < 8; ++i) {
    int r = rowbase + mb + i;
    if (r < N) {
      int c0 = colbase + n, c1 = c0 + 16;
      conv[(size_t)r * OUT_F + c0] =
          acc0[i] + bias[c0] + hskip[(size_t)r * IN_F + (c0 >> 1)];
      conv[(size_t)r * OUT_F + c1] =
          acc1[i] + bias[c1] + hskip[(size_t)r * IN_F + (c1 >> 1)];
    }
  }
}

// ---------------------------------------------------------------------------
// LayerNorm over 256 feats (in-place on conv) + bf16 copy; wave per row.
// ---------------------------------------------------------------------------
__global__ void k_ln2(float* __restrict__ x, const float* __restrict__ g,
                      const float* __restrict__ bta, __bf16* __restrict__ xb, int N) {
  int row  = (blockIdx.x * blockDim.x + threadIdx.x) >> 5;
  int lane = threadIdx.x & 31;
  if (row >= N) return;
  float* rp = x + (size_t)row * OUT_F + lane * 8;
  float4 v0 = *(const float4*)(rp);
  float4 v1 = *(const float4*)(rp + 4);
  float s  = v0.x + v0.y + v0.z + v0.w + v1.x + v1.y + v1.z + v1.w;
  float ss = v0.x*v0.x + v0.y*v0.y + v0.z*v0.z + v0.w*v0.w
           + v1.x*v1.x + v1.y*v1.y + v1.z*v1.z + v1.w*v1.w;
#pragma unroll
  for (int m = 16; m > 0; m >>= 1) {
    s  += __shfl_xor(s, m, 32);
    ss += __shfl_xor(ss, m, 32);
  }
  float mu = s * (1.0f / OUT_F);
  float rs = rsqrtf(ss * (1.0f / OUT_F) - mu * mu + 1e-5f);
  const float4 g0 = *(const float4*)(g + lane * 8);
  const float4 g1 = *(const float4*)(g + lane * 8 + 4);
  const float4 b0 = *(const float4*)(bta + lane * 8);
  const float4 b1 = *(const float4*)(bta + lane * 8 + 4);
  float4 o0, o1;
  o0.x = (v0.x - mu) * rs * g0.x + b0.x;
  o0.y = (v0.y - mu) * rs * g0.y + b0.y;
  o0.z = (v0.z - mu) * rs * g0.z + b0.z;
  o0.w = (v0.w - mu) * rs * g0.w + b0.w;
  o1.x = (v1.x - mu) * rs * g1.x + b1.x;
  o1.y = (v1.y - mu) * rs * g1.y + b1.y;
  o1.z = (v1.z - mu) * rs * g1.z + b1.z;
  o1.w = (v1.w - mu) * rs * g1.w + b1.w;
  *(float4*)(rp)     = o0;
  *(float4*)(rp + 4) = o1;
  v4bf p0 = { (__bf16)o0.x, (__bf16)o0.y, (__bf16)o0.z, (__bf16)o0.w };
  v4bf p1 = { (__bf16)o1.x, (__bf16)o1.y, (__bf16)o1.z, (__bf16)o1.w };
  *(v4bf*)(xb + (size_t)row * OUT_F + lane * 8)     = p0;
  *(v4bf*)(xb + (size_t)row * OUT_F + lane * 8 + 4) = p1;
}

// ---------------------------------------------------------------------------
// GEMM2: out = elu(x_bf16 @ W_si + b_si) + x_f32
// ---------------------------------------------------------------------------
__global__ void k_gemm2(const __bf16* __restrict__ X, const __bf16* __restrict__ W,
                        const float* __restrict__ bias, const float* __restrict__ xf,
                        float* __restrict__ out, int N) {
  const int LDA = OUT_F + 8;  // 264
  __shared__ __bf16 sA[16 * (OUT_F + 8)];
  const int tid = threadIdx.x;
  const int rowbase = blockIdx.x * 16;
  {
    int r  = tid >> 4;
    int c  = (tid & 15) * 16;
    int gr = rowbase + r; if (gr >= N) gr = N - 1;
    const uint4* sp = (const uint4*)(X + (size_t)gr * OUT_F + c);
    uint4* dp = (uint4*)&sA[r * LDA + c];
    dp[0] = sp[0];
    dp[1] = sp[1];
  }
  __syncthreads();
  const int lane    = tid & 31;
  const int colbase = (tid >> 5) * 32;
  const int arow  = lane & 15;
  const int khalf = (lane >> 4) * 8;
  const int bn    = lane & 15;
  const int bk    = (lane >> 4) * 16;
  v8f acc0 = {};
  v8f acc1 = {};
#pragma unroll
  for (int kt = 0; kt < OUT_F / 32; ++kt) {
    v16bf a = load_a_frag(&sA[arow * LDA + kt * 32 + khalf]);
    const __bf16* w = W + (((size_t)kt * OUT_F + colbase + bn) << 5) + bk;
    acc0 = wmma_bf16(a, *(const v16bf*)w, acc0);
    acc1 = wmma_bf16(a, *(const v16bf*)(w + (16 << 5)), acc1);
  }
  const int mb = (lane >> 4) * 8;
  const int n  = lane & 15;
#pragma unroll
  for (int i = 0; i < 8; ++i) {
    int r = rowbase + mb + i;
    if (r < N) {
      int c0 = colbase + n, c1 = c0 + 16;
      float y0 = acc0[i] + bias[c0];
      float y1 = acc1[i] + bias[c1];
      float e0 = y0 > 0.0f ? y0 : expm1f(y0);
      float e1 = y1 > 0.0f ? y1 : expm1f(y1);
      out[(size_t)r * OUT_F + c0] = e0 + xf[(size_t)r * OUT_F + c0];
      out[(size_t)r * OUT_F + c1] = e1 + xf[(size_t)r * OUT_F + c1];
    }
  }
}

// ---------------------------------------------------------------------------
extern "C" void kernel_launch(void* const* d_in, const int* in_sizes, int n_in,
                              void* d_out, int out_size, void* d_ws, size_t ws_size,
                              hipStream_t stream) {
  (void)n_in; (void)out_size; (void)ws_size;
  const float* h       = (const float*)d_in[0];
  const int*   src     = (const int*)d_in[1];
  const int*   dst     = (const int*)d_in[2];
  const float* ln1_g   = (const float*)d_in[3];
  const float* ln1_b   = (const float*)d_in[4];
  const float* W_self  = (const float*)d_in[5];
  const float* W_neigh = (const float*)d_in[6];
  const float* b_neigh = (const float*)d_in[7];
  const float* ln2_g   = (const float*)d_in[8];
  const float* ln2_b   = (const float*)d_in[9];
  const float* W_si    = (const float*)d_in[10];
  const float* b_si    = (const float*)d_in[11];
  const int N = in_sizes[0] / IN_F;
  const int E = in_sizes[1];
  float* out = (float*)d_out;

  char* w = (char*)d_ws;
  size_t off = 0;
  auto take = [&](size_t bytes) -> char* {
    char* p = w + off;
    off = (off + bytes + 255) & ~(size_t)255;
    return p;
  };
  float*  hn_f   = (float*)take((size_t)N * IN_F * 4);
  __bf16* hn_b   = (__bf16*)take((size_t)N * IN_F * 2);
  float*  agg    = (float*)take((size_t)N * IN_F * 4);
  float*  deg    = (float*)take((size_t)N * 4);
  __bf16* mean_b = (__bf16*)take((size_t)N * IN_F * 2);
  float*  conv   = (float*)take((size_t)N * OUT_F * 4);  // re-used as x_f32 (LN2 in place)
  __bf16* x_b    = (__bf16*)take((size_t)N * OUT_F * 2);
  __bf16* Wsp    = (__bf16*)take((size_t)IN_F * OUT_F * 2);
  __bf16* Wnp    = (__bf16*)take((size_t)IN_F * OUT_F * 2);
  __bf16* Wip    = (__bf16*)take((size_t)OUT_F * OUT_F * 2);

  hipMemsetAsync(agg, 0, (size_t)N * IN_F * 4, stream);
  hipMemsetAsync(deg, 0, (size_t)N * 4, stream);

  k_pack_w<<<(IN_F * OUT_F + 255) / 256, 256, 0, stream>>>(W_self, Wsp, IN_F);
  k_pack_w<<<(IN_F * OUT_F + 255) / 256, 256, 0, stream>>>(W_neigh, Wnp, IN_F);
  k_pack_w<<<(OUT_F * OUT_F + 255) / 256, 256, 0, stream>>>(W_si, Wip, OUT_F);

  k_ln1<<<(N + 7) / 8, 256, 0, stream>>>(h, ln1_g, ln1_b, hn_f, hn_b, N);
  k_scatter<<<(E + 7) / 8, 256, 0, stream>>>(hn_f, src, dst, agg, deg, E);
  k_mean<<<((size_t)N * 32 + 255) / 256, 256, 0, stream>>>(agg, deg, mean_b, N);
  k_gemm1<<<(N + 15) / 16, 256, 0, stream>>>(hn_b, mean_b, Wsp, Wnp, b_neigh, hn_f, conv, N);
  k_ln2<<<(N + 7) / 8, 256, 0, stream>>>(conv, ln2_g, ln2_b, x_b, N);
  k_gemm2<<<(N + 15) / 16, 256, 0, stream>>>(x_b, Wip, b_si, conv, out, N);
}
